// GANet_13718125543607
// MI455X (gfx1250) — matrix operand order repaired and checked by reading, not compiled, not run on hardware
//
#include <hip/hip_runtime.h>
#include <hip/hip_bf16.h>

// ---------------- problem constants ----------------
#define C_    32
#define D_    48
#define H_    64
#define W_    128
#define HW_   (H_ * W_)          // 8192
#define DHW_  (D_ * HW_)         // 393216
#define NTOT_ (C_ * DHW_)        // 12582912
#define NB_   96                 // partial-reduction blocks per channel
#define CHUNK_ (DHW_ / NB_)      // 4096

// padded LDS row strides (80 B = 20 dwords; gcd(20,64)=4 -> the 16 lanes of a
// WMMA fragment ds_load_b128 cover disjoint bank ranges; still 16B-aligned)
#define PPAD_ 40                 // patch row stride (elements)
#define WPAD_ 40                 // weight row stride (elements)

typedef __bf16 v16bf __attribute__((ext_vector_type(16)));
typedef __bf16 v8bf  __attribute__((ext_vector_type(8)));
typedef float  v8f   __attribute__((ext_vector_type(8)));

// =====================================================================
// SGA directional scan. One 64-thread block per (c, fixed) plane.
// Threads cover the D=48 depth column; recurrence marches along W or H.
// prev column kept in LDS (zero halo) for the d-1/d+1 taps; max over D
// via wave32 shfl_xor reduce + 2-entry LDS combine (parity-buffered).
// dir: 0 = W fwd, 1 = W rev, 2 = H fwd, 3 = H rev.
// dir 0 writes y; dirs 1..3 max-combine into y (stream-serialized).
// =====================================================================
__global__ __launch_bounds__(64) void sga_dir_kernel(
    const float* __restrict__ x, const float* __restrict__ g,
    float* __restrict__ y, int dir)
{
  const int tid  = threadIdx.x;
  const int d    = tid;
  const bool active = d < D_;
  const int lane = tid & 31;
  const int wid  = tid >> 5;

  int c, fix, L, stride;
  if (dir < 2) { c = blockIdx.x / H_; fix = blockIdx.x % H_; L = W_; stride = 1; }
  else         { c = blockIdx.x / W_; fix = blockIdx.x % W_; L = H_; stride = W_; }
  const bool rev  = (dir == 1) || (dir == 3);
  const int  gofs = dir * 160;   // k1..k4 channel blocks inside g

  long xbase, gbase;
  if (dir < 2) { xbase = (long)(c * D_ + d) * HW_ + (long)fix * W_;
                 gbase = (long)(gofs + c * 5) * HW_ + (long)fix * W_; }
  else         { xbase = (long)(c * D_ + d) * HW_ + fix;
                 gbase = (long)(gofs + c * 5) * HW_ + fix; }

  __shared__ float buf[D_ + 2];   // zero halo at [0] and [D_+1]
  __shared__ float wred[2][2];
  if (tid == 0) { buf[0] = 0.f; buf[D_ + 1] = 0.f; }

  // t = 0: out0 = x slice (reference: scan starts from xs[1:])
  const int pos0 = rev ? (L - 1) : 0;
  float out = 0.f;
  if (active) {
    long xi = xbase + (long)pos0 * stride;
    out = x[xi];
    if (dir == 0) y[xi] = out;
    else          y[xi] = fmaxf(y[xi], out);
  }
  float v = active ? out : -3.4e38f;
  #pragma unroll
  for (int o = 16; o >= 1; o >>= 1) v = fmaxf(v, __shfl_xor(v, o, 32));
  if (lane == 0) wred[0][wid] = v;
  __syncthreads();
  if (active) buf[1 + d] = out;
  __syncthreads();
  float pmax = fmaxf(wred[0][0], wred[0][1]);

  for (int t = 1; t < L; ++t) {
    const int pos = rev ? (L - 1 - t) : t;
    const long go = gbase + (long)pos * stride;
    // 5 raw weights; L1-normalize on the fly (shared denominator)
    float w0 = g[go];
    float w1 = g[go + (long)HW_];
    float w2 = g[go + (long)2 * HW_];
    float w3 = g[go + (long)3 * HW_];
    float w4 = g[go + (long)4 * HW_];
    float denom = fmaxf(fabsf(w0) + fabsf(w1) + fabsf(w2) + fabsf(w3) + fabsf(w4),
                        1e-12f);
    const long xi = xbase + (long)pos * stride;
    if (active) {
      float xt = x[xi];
      out = (w0 * xt + w1 * buf[d] + w2 * buf[d + 1] + w3 * buf[d + 2]
             + w4 * pmax) / denom;
      if (dir == 0) y[xi] = out;
      else          y[xi] = fmaxf(y[xi], out);
    }
    v = active ? out : -3.4e38f;
    #pragma unroll
    for (int o = 16; o >= 1; o >>= 1) v = fmaxf(v, __shfl_xor(v, o, 32));
    if (lane == 0) wred[t & 1][wid] = v;
    __syncthreads();               // all taps of old buf consumed; wred visible
    if (active) buf[1 + d] = out;
    __syncthreads();               // new buf visible
    pmax = fmaxf(wred[t & 1][0], wred[t & 1][1]);
  }
}

// =====================================================================
// Deterministic per-channel mean/var (two stage). No float atomics.
// =====================================================================
__global__ __launch_bounds__(256) void chan_stats_partial(
    const float* __restrict__ v, float* __restrict__ parts)
{
  const int c = blockIdx.x / NB_;
  const int b = blockIdx.x % NB_;
  const long base = (long)c * DHW_ + (long)b * CHUNK_;
  float s = 0.f, q = 0.f;
  #pragma unroll 4
  for (int i = threadIdx.x; i < CHUNK_; i += 256) {
    float t = v[base + i];
    s += t; q += t * t;
  }
  __shared__ float sh0[256], sh1[256];
  sh0[threadIdx.x] = s; sh1[threadIdx.x] = q;
  __syncthreads();
  for (int o = 128; o >= 1; o >>= 1) {
    if (threadIdx.x < o) {
      sh0[threadIdx.x] += sh0[threadIdx.x + o];
      sh1[threadIdx.x] += sh1[threadIdx.x + o];
    }
    __syncthreads();
  }
  if (threadIdx.x == 0) {
    parts[(c * NB_ + b) * 2 + 0] = sh0[0];
    parts[(c * NB_ + b) * 2 + 1] = sh1[0];
  }
}

__global__ __launch_bounds__(128) void chan_stats_final(
    const float* __restrict__ parts, const float* __restrict__ gamma,
    const float* __restrict__ beta, float* __restrict__ sb)
{
  const int c = blockIdx.x;
  float s = 0.f, q = 0.f;
  for (int i = threadIdx.x; i < NB_; i += 128) {
    s += parts[(c * NB_ + i) * 2 + 0];
    q += parts[(c * NB_ + i) * 2 + 1];
  }
  __shared__ float sh0[128], sh1[128];
  sh0[threadIdx.x] = s; sh1[threadIdx.x] = q;
  __syncthreads();
  for (int o = 64; o >= 1; o >>= 1) {
    if (threadIdx.x < o) {
      sh0[threadIdx.x] += sh0[threadIdx.x + o];
      sh1[threadIdx.x] += sh1[threadIdx.x + o];
    }
    __syncthreads();
  }
  if (threadIdx.x == 0) {
    float mean = sh0[0] / (float)DHW_;
    float var  = sh1[0] / (float)DHW_ - mean * mean;
    float inv  = rsqrtf(var + 1e-5f);
    float sc   = gamma[c] * inv;
    sb[c]      = sc;
    sb[C_ + c] = beta[c] - mean * sc;   // y*sc + bias == (y-mean)*inv*g + b
  }
}

// y = relu(y*scale + bias), in place
__global__ __launch_bounds__(256) void apply_bn_relu_kernel(
    float* __restrict__ y, const float* __restrict__ sb)
{
  int i = blockIdx.x * 256 + threadIdx.x;
  if (i >= NTOT_) return;
  int c = i / DHW_;
  y[i] = fmaxf(y[i] * sb[c] + sb[C_ + c], 0.f);
}

// out = relu(out*scale2 + bias2 + x), in place on d_out
__global__ __launch_bounds__(256) void final_kernel(
    float* __restrict__ out, const float* __restrict__ x,
    const float* __restrict__ sb)
{
  int i = blockIdx.x * 256 + threadIdx.x;
  if (i >= NTOT_) return;
  int c = i / DHW_;
  out[i] = fmaxf(out[i] * sb[c] + sb[C_ + c] + x[i], 0.f);
}

// =====================================================================
// 3x3x3 conv, 32->32 ch, implicit GEMM via v_wmma_f32_16x16x32_bf16.
// One block (8 waves) per (d,h) row. LDS: full weight tensor as bf16
// [tap][cout][cin·pad40] + input patch [dz][dy][wx][cin·pad40], cin
// innermost so each WMMA fragment is two conflict-free ds_load_b128.
// Patch staging iterates w fastest (coalesced 128B global reads).
// Per wave: M-tile = 16 contiguous w, two N-tiles (cout 0-15 / 16-31),
// K loop = 27 taps x 32 cin  =>  54 WMMAs per wave.
// =====================================================================
__global__ __launch_bounds__(256) void conv_wmma_kernel(
    const float* __restrict__ yin,  // bn1+relu'd activations, f32
    const float* __restrict__ cw,   // conv_w (O,I,3,3,3) f32
    float* __restrict__ out)        // f32 (C,D,H,W)
{
  __shared__ __align__(16) __bf16 wlds[27 * C_ * WPAD_];         // 69120 B
  __shared__ __align__(16) __bf16 patch[3 * 3 * (W_ + 2) * PPAD_]; // 93600 B
  const int PW = W_ + 2;

  const int tid = threadIdx.x;
  const int dd  = blockIdx.x / H_;
  const int hh  = blockIdx.x % H_;

  // stage weights: wlds[(tap*32 + o)*WPAD_ + i] = cw[o][i][tap]
  // (ci fastest -> conflict-free ds stores; global side is tiny/L2-hot)
  #pragma unroll 4
  for (int i = tid; i < 27 * C_ * C_; i += 256) {
    int ci  = i & 31;
    int o   = (i >> 5) & 31;
    int tap = i >> 10;
    wlds[(tap * C_ + o) * WPAD_ + ci] = (__bf16)cw[(o * C_ + ci) * 27 + tap];
  }
  // stage patch, w fastest for coalesced global reads; zero-pad borders
  #pragma unroll 2
  for (int j = tid; j < 3 * 3 * PW * C_; j += 256) {
    int wx = j % PW;
    int r  = j / PW;        // r = (dz*3+dy)*32 + ci
    int ci = r & 31;
    int q  = r >> 5;
    int dy = q % 3;
    int dz = q / 3;
    int zd = dd + dz - 1;
    int yh = hh + dy - 1;
    int xw = wx - 1;
    float val = 0.f;
    if (zd >= 0 && zd < D_ && yh >= 0 && yh < H_ && xw >= 0 && xw < W_)
      val = yin[(long)(ci * D_ + zd) * HW_ + yh * W_ + xw];
    patch[((q * PW) + wx) * PPAD_ + ci] = (__bf16)val;
  }
  __syncthreads();

  const int wv    = tid >> 5;
  const int lane  = tid & 31;
  const int ln16  = lane & 15;
  const int half  = lane >> 4;
  const int w0seg = wv * 16;

  v8f acc0 = {};
  v8f acc1 = {};

  for (int tap = 0; tap < 27; ++tap) {
    const int dz = tap / 9, dy = (tap / 3) % 3, dx = tap % 3;
    // A fragment (16x32 bf16): lane row M = ln16; K = cin.
    // ISA layout: e in [0,7] -> K = half*8 + e ; e in [8,15] -> K = 16 + half*8 + (e-8)
    const v8bf* prow =
        (const v8bf*)&patch[(((dz * 3 + dy) * PW) + (w0seg + ln16 + dx)) * PPAD_];
    v8bf alo = prow[half];       // cin  8*half .. 8*half+7
    v8bf ahi = prow[2 + half];   // cin 16+8*half .. +7
    v16bf A;
    #pragma unroll
    for (int e = 0; e < 8; ++e) { A[e] = alo[e]; A[8 + e] = ahi[e]; }

    // B fragments (32x16 bf16): lane col N = ln16 (+16); K same order as A.
    const v8bf* b0p = (const v8bf*)&wlds[(tap * C_ + ln16) * WPAD_];
    v8bf b0a = b0p[half], b0b = b0p[2 + half];
    const v8bf* b1p = (const v8bf*)&wlds[(tap * C_ + 16 + ln16) * WPAD_];
    v8bf b1a = b1p[half], b1b = b1p[2 + half];
    v16bf B0, B1;
    #pragma unroll
    for (int e = 0; e < 8; ++e) {
      B0[e] = b0a[e]; B0[8 + e] = b0b[e];
      B1[e] = b1a[e]; B1[8 + e] = b1b[e];
    }

    acc0 = __builtin_amdgcn_wmma_f32_16x16x32_bf16(false, A, false, B0,
                                                   (short)0, acc0, false, false);
    acc1 = __builtin_amdgcn_wmma_f32_16x16x32_bf16(false, A, false, B1,
                                                   (short)0, acc1, false, false);
  }

  // D layout: reg r -> M = half*8 + r (spatial w), N = ln16 (cout)
  // each lane owns 8 consecutive w -> two b128 stores per tile
  const int wbase = w0seg + half * 8;
  float* p0 = out + (long)(ln16 * D_ + dd) * HW_ + hh * W_ + wbase;
  float* p1 = out + (long)((ln16 + 16) * D_ + dd) * HW_ + hh * W_ + wbase;
  float4 s0a = make_float4(acc0[0], acc0[1], acc0[2], acc0[3]);
  float4 s0b = make_float4(acc0[4], acc0[5], acc0[6], acc0[7]);
  float4 s1a = make_float4(acc1[0], acc1[1], acc1[2], acc1[3]);
  float4 s1b = make_float4(acc1[4], acc1[5], acc1[6], acc1[7]);
  *(float4*)(p0)     = s0a;
  *(float4*)(p0 + 4) = s0b;
  *(float4*)(p1)     = s1a;
  *(float4*)(p1 + 4) = s1b;
}

// =====================================================================
extern "C" void kernel_launch(void* const* d_in, const int* in_sizes, int n_in,
                              void* d_out, int out_size, void* d_ws, size_t ws_size,
                              hipStream_t stream) {
  const float* x  = (const float*)d_in[0];
  const float* g  = (const float*)d_in[1];
  const float* cw = (const float*)d_in[2];
  const float* g1 = (const float*)d_in[3];
  const float* b1 = (const float*)d_in[4];
  const float* g2 = (const float*)d_in[5];
  const float* b2 = (const float*)d_in[6];
  float* out = (float*)d_out;

  char*  ws    = (char*)d_ws;
  float* y     = (float*)ws;                              // NTOT_ f32 (50 MB)
  float* parts = (float*)(ws + (size_t)NTOT_ * 4);        // C_*NB_*2 f32
  float* sb1   = parts + C_ * NB_ * 2;                    // 2*C_ f32
  float* sb2   = sb1 + 2 * C_;                            // 2*C_ f32

  // 1) SGA: 4 directional scans, max-combined into y
  sga_dir_kernel<<<C_ * H_, 64, 0, stream>>>(x, g, y, 0);
  sga_dir_kernel<<<C_ * H_, 64, 0, stream>>>(x, g, y, 1);
  sga_dir_kernel<<<C_ * W_, 64, 0, stream>>>(x, g, y, 2);
  sga_dir_kernel<<<C_ * W_, 64, 0, stream>>>(x, g, y, 3);

  // 2) BN1 stats + relu(bn1(y)) in place
  chan_stats_partial<<<C_ * NB_, 256, 0, stream>>>(y, parts);
  chan_stats_final<<<C_, 128, 0, stream>>>(parts, g1, b1, sb1);
  apply_bn_relu_kernel<<<(NTOT_ + 255) / 256, 256, 0, stream>>>(y, sb1);

  // 3) conv3d via bf16 WMMA -> d_out (pre-bn2)
  conv_wmma_kernel<<<D_ * H_, 256, 0, stream>>>(y, cw, out);

  // 4) BN2 stats + fused bn2 + residual + relu in place on d_out
  chan_stats_partial<<<C_ * NB_, 256, 0, stream>>>(out, parts);
  chan_stats_final<<<C_, 128, 0, stream>>>(parts, g2, b2, sb2);
  final_kernel<<<(NTOT_ + 255) / 256, 256, 0, stream>>>(out, x, sb2);
}
